// EncoderLayer_68315749810849
// MI455X (gfx1250) — compile-verified, matrix operand order checked by reading, not run on hardware
//
#include <hip/hip_runtime.h>
#include <hip/hip_bf16.h>

// ---------------------------------------------------------------------------
// Feature probes (evaluated per compile pass; host pass falls back cleanly)
// ---------------------------------------------------------------------------
#if defined(__has_builtin)
#  if __has_builtin(__builtin_amdgcn_global_load_async_to_lds_b128) && \
      __has_builtin(__builtin_amdgcn_s_wait_asynccnt)
#    define USE_ASYNC 1
#  endif
#  if __has_builtin(__builtin_amdgcn_tensor_load_to_lds) && \
      __has_builtin(__builtin_amdgcn_s_wait_tensorcnt)
#    define USE_TDM 1
#  endif
#endif
#ifndef USE_ASYNC
#define USE_ASYNC 0
#endif
#ifndef USE_TDM
#define USE_TDM 0
#endif

#if USE_TDM
#define ASYNC_PER_STAGE 1   // per wave per stage: 1 async (A tile); W via TDM
#else
#define ASYNC_PER_STAGE 5   // 1 (A tile) + 4 (W tile)
#endif

// ---------------------------------------------------------------------------
// Types
// ---------------------------------------------------------------------------
typedef __attribute__((ext_vector_type(16))) __bf16 v16bf;
typedef __attribute__((ext_vector_type(8)))  float  v8f;
typedef __attribute__((ext_vector_type(4)))  unsigned int u32x4;
typedef __attribute__((ext_vector_type(8)))  int i32x8;
typedef __attribute__((ext_vector_type(4)))  int i32x4;
typedef __attribute__((ext_vector_type(4)))  int v4i;
// pointer-to-v4i in global (AS1) / LDS (AS3) address spaces
typedef __attribute__((address_space(1))) v4i* gv4i_p;
typedef __attribute__((address_space(3))) v4i* lv4i_p;

union Frag {
    v16bf v;
    u32x4 u4[2];
};

__device__ __forceinline__ unsigned short f32_to_bf16(float f) {
    unsigned int u = __float_as_uint(f);
    unsigned int r = (u + 0x7FFFu + ((u >> 16) & 1u)) >> 16;
    return (unsigned short)r;
}

__device__ __forceinline__ v8f wmma_bf16(v16bf a, v16bf b, v8f c) {
    // D = A(16x32 bf16) x B(32x16 bf16) + C(16x16 f32)
    return __builtin_amdgcn_wmma_f32_16x16x32_bf16(
        /*neg_a=*/false, a, /*neg_b=*/false, b,
        /*c_mod=*/(short)0, c, /*reuse_a=*/false, /*reuse_b=*/false);
}

// ---------------------------------------------------------------------------
// Elementwise casts / transposes
// ---------------------------------------------------------------------------
__global__ void cast_bf16_kernel(const float* __restrict__ in,
                                 unsigned short* __restrict__ out, int n) {
    int i = blockIdx.x * blockDim.x + threadIdx.x;
    if (i < n) out[i] = f32_to_bf16(in[i]);
}

// W[K][N] fp32 -> Wt[N][K] bf16
__global__ void cast_transpose_kernel(const float* __restrict__ W,
                                      unsigned short* __restrict__ Wt,
                                      int K, int N) {
    int i = blockIdx.x * blockDim.x + threadIdx.x;
    if (i < K * N) {
        int k = i % K;
        int n = i / K;
        Wt[i] = f32_to_bf16(W[(size_t)k * N + n]);
    }
}

// v[b,s,h,dk] bf16 -> vt[b,h,dk,s] bf16
__global__ void transpose_v_kernel(const unsigned short* __restrict__ v,
                                   unsigned short* __restrict__ vt, int total) {
    const int S = 2048;
    int i = blockIdx.x * blockDim.x + threadIdx.x;
    if (i >= total) return;
    int s = i & (S - 1);
    int r = i >> 11;          // (b*16 + h)*64 + dk
    int dk = r & 63; r >>= 6; // b*16 + h
    int h  = r & 15;
    int b  = r >> 4;
    vt[i] = v[(((size_t)(b * S + s)) * 16 + h) * 64 + dk];
}

// ---------------------------------------------------------------------------
// gate[b,s,h] = sigmoid(x[b,s,:] . w_g[:,h] + b_g[h])
// ---------------------------------------------------------------------------
__global__ void gate_kernel(const float* __restrict__ x,
                            const float* __restrict__ wg,
                            const float* __restrict__ bg,
                            float* __restrict__ gate, int rows) {
    const int D = 1024, H = 16;
    int gid = blockIdx.x * blockDim.x + threadIdx.x;
    if (gid >= rows * H) return;
    int row = gid >> 4;
    int h   = gid & 15;
    const float* xr = x + (size_t)row * D;
    float acc = bg[h];
#pragma unroll 4
    for (int d = 0; d < D; ++d) acc += xr[d] * wg[d * H + h];
    gate[gid] = 1.0f / (1.0f + __expf(-acc));
}

// ---------------------------------------------------------------------------
// GEMM tile staging: A tile (64x32) via async-to-LDS, W tile (256x32) via TDM.
// LDS rows padded to 40 ushorts (TDM pad: every 16 DWORDs insert 4 DWORDs).
// ---------------------------------------------------------------------------
__device__ __forceinline__ void gemm_stage(
    const unsigned short* __restrict__ A, const unsigned short* __restrict__ Wt,
    unsigned short* Asb, unsigned short* Wsb,
    int m0, int n0, int K, int k0, int arow, int ach, int tid, int wv) {
#if USE_ASYNC
    __builtin_amdgcn_global_load_async_to_lds_b128(
        (gv4i_p)(const void*)(A + (size_t)(m0 + arow) * K + k0 + ach),
        (lv4i_p)(void*)(Asb + arow * 40 + ach), 0, 0);
#else
    *(u32x4*)(Asb + arow * 40 + ach) =
        *(const u32x4*)(A + (size_t)(m0 + arow) * K + k0 + ach);
#endif

#if USE_TDM
    if (wv == 0) {
        unsigned long long ga =
            (unsigned long long)(size_t)(Wt + (size_t)n0 * K + k0);
        unsigned ldsoff = (unsigned)(size_t)(void*)Wsb;
        u32x4 g0;
        g0[0] = 1u;             // count=1 (valid descriptor)
        g0[1] = ldsoff;         // lds_addr
        g0[2] = (unsigned)ga;   // global_addr[31:0]
        g0[3] = (unsigned)((ga >> 32) & 0x1FFFFFFull) | 0x80000000u; // hi|type=2
        i32x8 g1;
        g1[0] = (1 << 16) | (1 << 20) | (3 << 22) | (3 << 25);
        //        data=2B    pad_en     every 16DW   pad 4DW
        g1[1] = (K & 0xFFFF) << 16;                    // tensor_dim0 lo16
        g1[2] = ((unsigned)K >> 16) | (256 << 16);     // dim0 hi | dim1 lo (=256)
        g1[3] = (32 << 16);                            // dim1 hi=0 | tile_dim0=32
        g1[4] = 256;                                   // tile_dim1=256, tile_dim2=0
        g1[5] = K;                                     // tensor_dim0_stride lo
        g1[6] = 0;
        g1[7] = 0;
        i32x4 gz;
        gz[0] = 0; gz[1] = 0; gz[2] = 0; gz[3] = 0;
        i32x8 gz8;
        gz8[0] = 0; gz8[1] = 0; gz8[2] = 0; gz8[3] = 0;
        gz8[4] = 0; gz8[5] = 0; gz8[6] = 0; gz8[7] = 0;
        __builtin_amdgcn_tensor_load_to_lds(g0, g1, gz, gz, gz8, 0);
    }
#elif USE_ASYNC
#pragma unroll
    for (int j = 0; j < 4; ++j) {
        int c = tid + 256 * j;
        int wr = c >> 2, wc = (c & 3) * 8;
        __builtin_amdgcn_global_load_async_to_lds_b128(
            (gv4i_p)(const void*)(Wt + (size_t)(n0 + wr) * K + k0 + wc),
            (lv4i_p)(void*)(Wsb + wr * 40 + wc), 0, 0);
    }
#else
#pragma unroll
    for (int j = 0; j < 4; ++j) {
        int c = tid + 256 * j;
        int wr = c >> 2, wc = (c & 3) * 8;
        *(u32x4*)(Wsb + wr * 40 + wc) =
            *(const u32x4*)(Wt + (size_t)(n0 + wr) * K + k0 + wc);
    }
#endif
}

// ---------------------------------------------------------------------------
// WMMA GEMM: out = act(A[M,K](bf16) @ Wt[N,K]^T(bf16) + bias[N])
// block: 256 threads (8 waves), tile 64(M) x 256(N), K-step 32, double buffer
// wave tile 32x64 -> 8 WMMAs per K-step per wave
// ---------------------------------------------------------------------------
template <int ACT, bool WF32, bool WB16>
__global__ __launch_bounds__(256) void gemm_bf16_kernel(
    const unsigned short* __restrict__ A, const unsigned short* __restrict__ Wt,
    const float* __restrict__ bias, float* __restrict__ outF,
    unsigned short* __restrict__ outB, int M, int N, int K) {
    const int AST = 40, WST = 40;
    __shared__ unsigned short As[2][64 * 40];
    __shared__ unsigned short Ws[2][256 * 40];

    int tid  = threadIdx.x;
    int lane = tid & 31;
    int wv   = tid >> 5;
    int ln16 = lane & 15;
    int half = lane >> 4;
    int kb   = half * 8;

    int m0 = blockIdx.y * 64;
    int n0 = blockIdx.x * 256;
    int mw = (wv >> 2) * 32;   // 2 wave rows of 32
    int nw = (wv & 3) * 64;    // 4 wave cols of 64

    v8f acc[2][4];
#pragma unroll
    for (int mi = 0; mi < 2; ++mi)
#pragma unroll
        for (int ni = 0; ni < 4; ++ni)
#pragma unroll
            for (int e = 0; e < 8; ++e) acc[mi][ni][e] = 0.0f;

    int arow = tid >> 2, ach = (tid & 3) * 8; // A tile: 64 rows x 4 chunks

    gemm_stage(A, Wt, As[0], Ws[0], m0, n0, K, 0, arow, ach, tid, wv);

    int nIter = K >> 5;
    for (int it = 0; it < nIter; ++it) {
        int buf = it & 1;
        bool more = (it + 1 < nIter);
        if (more)
            gemm_stage(A, Wt, As[buf ^ 1], Ws[buf ^ 1], m0, n0, K,
                       (it + 1) << 5, arow, ach, tid, wv);
#if USE_ASYNC
        if (more) __builtin_amdgcn_s_wait_asynccnt(ASYNC_PER_STAGE);
        else      __builtin_amdgcn_s_wait_asynccnt(0);
#endif
#if USE_TDM
        if (wv == 0) {
            if (more) __builtin_amdgcn_s_wait_tensorcnt(1);
            else      __builtin_amdgcn_s_wait_tensorcnt(0);
        }
#endif
        __syncthreads();

        const unsigned short* Asb = As[buf];
        const unsigned short* Wsb = Ws[buf];
        Frag af[2], bf[4];
#pragma unroll
        for (int mi = 0; mi < 2; ++mi) {
            int r = mw + mi * 16 + ln16;
            af[mi].u4[0] = *(const u32x4*)&Asb[r * AST + kb];
            af[mi].u4[1] = *(const u32x4*)&Asb[r * AST + 16 + kb];
        }
#pragma unroll
        for (int ni = 0; ni < 4; ++ni) {
            int c = nw + ni * 16 + ln16;
            bf[ni].u4[0] = *(const u32x4*)&Wsb[c * WST + kb];
            bf[ni].u4[1] = *(const u32x4*)&Wsb[c * WST + 16 + kb];
        }
#pragma unroll
        for (int mi = 0; mi < 2; ++mi)
#pragma unroll
            for (int ni = 0; ni < 4; ++ni)
                acc[mi][ni] = wmma_bf16(af[mi].v, bf[ni].v, acc[mi][ni]);
        __syncthreads();
    }

#pragma unroll
    for (int mi = 0; mi < 2; ++mi)
#pragma unroll
        for (int ni = 0; ni < 4; ++ni) {
            int col = n0 + nw + ni * 16 + ln16;
            float bv = bias[col];
#pragma unroll
            for (int v = 0; v < 8; ++v) {
                int row = m0 + mw + mi * 16 + v + 8 * half;
                float val = acc[mi][ni][v] + bv;
                if (ACT == 1)
                    val = 0.5f * val * (1.0f + erff(val * 0.70710678118654752f));
                if (WF32) outF[(size_t)row * N + col] = val;
                if (WB16) outB[(size_t)row * N + col] = f32_to_bf16(val);
            }
        }
}

// ---------------------------------------------------------------------------
// Flash attention (+ post-softmax sigmoid gate folded into epilogue)
// grid: (S/64, H, B), block 128 = 4 waves, each wave owns a 16-query tile.
// ---------------------------------------------------------------------------
__global__ __launch_bounds__(128) void flash_attn_kernel(
    const unsigned short* __restrict__ qg, const unsigned short* __restrict__ kg,
    const unsigned short* __restrict__ vt, const float* __restrict__ gate,
    const int* __restrict__ mask, unsigned short* __restrict__ ctx) {
    const int S = 2048, H = 16;
    int tid  = threadIdx.x;
    int w    = tid >> 5;
    int lane = tid & 31;
    int ln16 = lane & 15;
    int half = lane >> 4;
    int kb   = half * 8;

    int b  = blockIdx.z;
    int h  = blockIdx.y;
    int q0 = blockIdx.x * 64 + w * 16;

    __shared__ unsigned short Plds[4][16 * 32];
    unsigned short* P = Plds[w];

    const unsigned short* qbase = qg + (size_t)b * S * 1024 + h * 64;
    const unsigned short* kbase = kg + (size_t)b * S * 1024 + h * 64;
    const unsigned short* vbase = vt + (size_t)(b * H + h) * 64 * S;
    const int* mrow = mask + b * S;

    Frag qlo, qhi;
    {
        const unsigned short* qr = qbase + (size_t)(q0 + ln16) * 1024;
        qlo.u4[0] = *(const u32x4*)(qr + kb);
        qlo.u4[1] = *(const u32x4*)(qr + 16 + kb);
        qhi.u4[0] = *(const u32x4*)(qr + 32 + kb);
        qhi.u4[1] = *(const u32x4*)(qr + 48 + kb);
    }

    v8f ctxa[4];
    float mrun[8], lrun[8];
#pragma unroll
    for (int g = 0; g < 4; ++g)
#pragma unroll
        for (int e = 0; e < 8; ++e) ctxa[g][e] = 0.0f;
#pragma unroll
    for (int v = 0; v < 8; ++v) { mrun[v] = -1e30f; lrun[v] = 0.0f; }

    for (int kt = 0; kt < S; kt += 32) {
        Frag k0lo, k0hi, k1lo, k1hi;
        const unsigned short* kr0 = kbase + (size_t)(kt + ln16) * 1024;
        const unsigned short* kr1 = kbase + (size_t)(kt + 16 + ln16) * 1024;
        k0lo.u4[0] = *(const u32x4*)(kr0 + kb);
        k0lo.u4[1] = *(const u32x4*)(kr0 + 16 + kb);
        k0hi.u4[0] = *(const u32x4*)(kr0 + 32 + kb);
        k0hi.u4[1] = *(const u32x4*)(kr0 + 48 + kb);
        k1lo.u4[0] = *(const u32x4*)(kr1 + kb);
        k1lo.u4[1] = *(const u32x4*)(kr1 + 16 + kb);
        k1hi.u4[0] = *(const u32x4*)(kr1 + 32 + kb);
        k1hi.u4[1] = *(const u32x4*)(kr1 + 48 + kb);

        if (kt + 32 < S) {
            __builtin_prefetch((const void*)(kr0 + 32 * 1024), 0, 0);
            __builtin_prefetch((const void*)(kr1 + 32 * 1024), 0, 0);
            __builtin_prefetch((const void*)(vbase + (size_t)ln16 * S + kt + 32), 0, 0);
        }

        v8f c0, c1;
#pragma unroll
        for (int e = 0; e < 8; ++e) { c0[e] = 0.0f; c1[e] = 0.0f; }
        c0 = wmma_bf16(qlo.v, k0lo.v, c0);
        c0 = wmma_bf16(qhi.v, k0hi.v, c0);
        c1 = wmma_bf16(qlo.v, k1lo.v, c1);
        c1 = wmma_bf16(qhi.v, k1hi.v, c1);

        int mv0 = mrow[kt + ln16];
        int mv1 = mrow[kt + 16 + ln16];

#pragma unroll
        for (int v = 0; v < 8; ++v) {
            float s0 = c0[v] * 0.125f;
            float s1 = c1[v] * 0.125f;
            s0 = mv0 ? s0 : -1e9f;
            s1 = mv1 ? s1 : -1e9f;
            float t = fmaxf(s0, s1);
            t = fmaxf(t, __shfl_xor(t, 1));
            t = fmaxf(t, __shfl_xor(t, 2));
            t = fmaxf(t, __shfl_xor(t, 4));
            t = fmaxf(t, __shfl_xor(t, 8));
            float mnew  = fmaxf(mrun[v], t);
            float alpha = __expf(mrun[v] - mnew);
            mrun[v] = mnew;
            float p0 = __expf(s0 - mnew);
            float p1 = __expf(s1 - mnew);
            float rs = p0 + p1;
            rs += __shfl_xor(rs, 1);
            rs += __shfl_xor(rs, 2);
            rs += __shfl_xor(rs, 4);
            rs += __shfl_xor(rs, 8);
            lrun[v] = lrun[v] * alpha + rs;
#pragma unroll
            for (int g = 0; g < 4; ++g) ctxa[g][v] *= alpha;
            int prow = (v + 8 * half) * 32;
            P[prow + ln16]      = f32_to_bf16(p0);
            P[prow + 16 + ln16] = f32_to_bf16(p1);
        }
        __syncthreads();

        Frag pf;
        pf.u4[0] = *(const u32x4*)&P[ln16 * 32 + kb];
        pf.u4[1] = *(const u32x4*)&P[ln16 * 32 + 16 + kb];

#pragma unroll
        for (int g = 0; g < 4; ++g) {
            Frag vf;
            const unsigned short* vr = vbase + (size_t)(16 * g + ln16) * S + kt;
            vf.u4[0] = *(const u32x4*)(vr + kb);
            vf.u4[1] = *(const u32x4*)(vr + 16 + kb);
            ctxa[g] = wmma_bf16(pf.v, vf.v, ctxa[g]);
        }
        __syncthreads();
    }

#pragma unroll
    for (int v = 0; v < 8; ++v) {
        int r = q0 + v + 8 * half;
        float gq  = gate[(size_t)(b * S + r) * H + h];
        float inv = gq / lrun[v];
#pragma unroll
        for (int g = 0; g < 4; ++g) {
            float val = ctxa[g][v] * inv;
            ctx[((size_t)(b * S + r) * H + h) * 64 + 16 * g + ln16] =
                f32_to_bf16(val);
        }
    }
}

// ---------------------------------------------------------------------------
// out = LayerNorm(a + r) * g + beta ; optional bf16 copy
// ---------------------------------------------------------------------------
__global__ __launch_bounds__(256) void add_ln_kernel(
    const float* __restrict__ a, const float* __restrict__ r,
    const float* __restrict__ g, const float* __restrict__ beta,
    float* __restrict__ outF, unsigned short* __restrict__ outB) {
    const int D = 1024;
    __shared__ float s1[256], s2[256];
    int row = blockIdx.x, tid = threadIdx.x;
    const float* ar = a + (size_t)row * D;
    const float* rr = r + (size_t)row * D;
    float vals[4], sum = 0.0f, sq = 0.0f;
#pragma unroll
    for (int i = 0; i < 4; ++i) {
        int c = i * 256 + tid;
        float v = ar[c] + rr[c];
        vals[i] = v;
        sum += v;
        sq += v * v;
    }
    s1[tid] = sum; s2[tid] = sq;
    __syncthreads();
    for (int st = 128; st > 0; st >>= 1) {
        if (tid < st) { s1[tid] += s1[tid + st]; s2[tid] += s2[tid + st]; }
        __syncthreads();
    }
    float mu  = s1[0] * (1.0f / D);
    float var = s2[0] * (1.0f / D) - mu * mu;
    float rs  = rsqrtf(var + 1e-5f);
#pragma unroll
    for (int i = 0; i < 4; ++i) {
        int c = i * 256 + tid;
        float o = (vals[i] - mu) * rs * g[c] + beta[c];
        outF[(size_t)row * D + c] = o;
        if (outB) outB[(size_t)row * D + c] = f32_to_bf16(o);
    }
}

// ---------------------------------------------------------------------------
// Host orchestration
// ---------------------------------------------------------------------------
static char* carve(void* base, size_t& off, size_t bytes) {
    char* p = (char*)base + off;
    off += (bytes + 255) & ~(size_t)255;
    return p;
}

extern "C" void kernel_launch(void* const* d_in, const int* in_sizes, int n_in,
                              void* d_out, int out_size, void* d_ws, size_t ws_size,
                              hipStream_t stream) {
    (void)in_sizes; (void)n_in; (void)out_size; (void)ws_size;
    const int Bb = 4, S = 2048, D = 1024, H = 16, DFF = 4096;
    const int M = Bb * S; // 8192

    const float* x    = (const float*)d_in[0];
    const float* w_q  = (const float*)d_in[1];  const float* b_q = (const float*)d_in[2];
    const float* w_k  = (const float*)d_in[3];  const float* b_k = (const float*)d_in[4];
    const float* w_v  = (const float*)d_in[5];  const float* b_v = (const float*)d_in[6];
    const float* w_o  = (const float*)d_in[7];  const float* b_o = (const float*)d_in[8];
    const float* w_g  = (const float*)d_in[9];  const float* b_g = (const float*)d_in[10];
    const float* w1   = (const float*)d_in[11]; const float* b1  = (const float*)d_in[12];
    const float* w2   = (const float*)d_in[13]; const float* b2  = (const float*)d_in[14];
    const float* g1   = (const float*)d_in[15]; const float* be1 = (const float*)d_in[16];
    const float* g2   = (const float*)d_in[17]; const float* be2 = (const float*)d_in[18];
    const int*   mask = (const int*)d_in[19];
    float* out = (float*)d_out;

    size_t off = 0;
    unsigned short* xbf  = (unsigned short*)carve(d_ws, off, (size_t)M * D * 2);
    unsigned short* wqt  = (unsigned short*)carve(d_ws, off, (size_t)D * D * 2);
    unsigned short* wkt  = (unsigned short*)carve(d_ws, off, (size_t)D * D * 2);
    unsigned short* wvt  = (unsigned short*)carve(d_ws, off, (size_t)D * D * 2);
    unsigned short* wot  = (unsigned short*)carve(d_ws, off, (size_t)D * D * 2);
    unsigned short* w1t  = (unsigned short*)carve(d_ws, off, (size_t)D * DFF * 2);
    unsigned short* w2t  = (unsigned short*)carve(d_ws, off, (size_t)DFF * D * 2);
    unsigned short* qbf  = (unsigned short*)carve(d_ws, off, (size_t)M * D * 2);
    unsigned short* kbf  = (unsigned short*)carve(d_ws, off, (size_t)M * D * 2);
    unsigned short* vbf  = (unsigned short*)carve(d_ws, off, (size_t)M * D * 2);
    unsigned short* vtbf = (unsigned short*)carve(d_ws, off, (size_t)M * D * 2);
    float*          gbuf = (float*)carve(d_ws, off, (size_t)M * H * 4);
    unsigned short* ctxb = (unsigned short*)carve(d_ws, off, (size_t)M * D * 2);
    float*          attn = (float*)carve(d_ws, off, (size_t)M * D * 4);
    float*          x1f  = (float*)carve(d_ws, off, (size_t)M * D * 4);
    unsigned short* x1b  = (unsigned short*)carve(d_ws, off, (size_t)M * D * 2);
    unsigned short* hbf  = (unsigned short*)carve(d_ws, off, (size_t)M * DFF * 2);
    float*          fff  = (float*)carve(d_ws, off, (size_t)M * D * 4);

    cast_bf16_kernel<<<(M * D + 255) / 256, 256, 0, stream>>>(x, xbf, M * D);
    cast_transpose_kernel<<<(D * D + 255) / 256, 256, 0, stream>>>(w_q, wqt, D, D);
    cast_transpose_kernel<<<(D * D + 255) / 256, 256, 0, stream>>>(w_k, wkt, D, D);
    cast_transpose_kernel<<<(D * D + 255) / 256, 256, 0, stream>>>(w_v, wvt, D, D);
    cast_transpose_kernel<<<(D * D + 255) / 256, 256, 0, stream>>>(w_o, wot, D, D);
    cast_transpose_kernel<<<(D * DFF + 255) / 256, 256, 0, stream>>>(w1, w1t, D, DFF);
    cast_transpose_kernel<<<(DFF * D + 255) / 256, 256, 0, stream>>>(w2, w2t, DFF, D);

    // QKV projections (bf16 out); block tile 64x256
    dim3 gD(D / 256, M / 64);
    gemm_bf16_kernel<0, false, true><<<gD, 256, 0, stream>>>(xbf, wqt, b_q, nullptr, qbf, M, D, D);
    gemm_bf16_kernel<0, false, true><<<gD, 256, 0, stream>>>(xbf, wkt, b_k, nullptr, kbf, M, D, D);
    gemm_bf16_kernel<0, false, true><<<gD, 256, 0, stream>>>(xbf, wvt, b_v, nullptr, vbf, M, D, D);

    transpose_v_kernel<<<(M * D + 255) / 256, 256, 0, stream>>>(vbf, vtbf, M * D);
    gate_kernel<<<(M * H + 255) / 256, 256, 0, stream>>>(x, w_g, b_g, gbuf, M);
    flash_attn_kernel<<<dim3(S / 64, H, Bb), 128, 0, stream>>>(qbf, kbf, vtbf, gbuf, mask, ctxb);

    gemm_bf16_kernel<0, true, false><<<gD, 256, 0, stream>>>(ctxb, wot, b_o, attn, nullptr, M, D, D);

    add_ln_kernel<<<M, 256, 0, stream>>>(x, attn, g1, be1, x1f, x1b);

    dim3 gF(DFF / 256, M / 64);
    gemm_bf16_kernel<1, false, true><<<gF, 256, 0, stream>>>(x1b, w1t, b1, nullptr, hbf, M, DFF, D);
    gemm_bf16_kernel<0, true, false><<<gD, 256, 0, stream>>>(hbf, w2t, b2, fff, nullptr, M, D, DFF);

    add_ln_kernel<<<M, 256, 0, stream>>>(x1f, fff, g2, be2, out, nullptr);
}
